// ProteinEnergyNet_63608465654249
// MI455X (gfx1250) — compile-verified, matrix operand order checked by reading, not run on hardware
//
#include <hip/hip_runtime.h>

// ---------------- constants ----------------
#define BATCH   4
#define NRES    512
#define EMBW    64
#define DFEAT   112      // 48 + 64
#define NFILT   128
#define NCH     48       // 3*NAD conv2d output channels
#define CIN     16       // NAD
#define KC2D    416      // conv2d K = 16*25 = 400 padded to 416 (13 steps)
#define KCV1    1248     // conv1 K = 112*11 = 1232 padded to 1248 (39 steps)
#define KCV2    1408     // conv2 K = 128*11 = 1408 exact (44 steps)
#define C1C     10.5f    // 0.5 + 1/H
#define C2C     (-9.5f)  // 0.5 - 1/H
#define ALPHA_C 0.1f

typedef __attribute__((ext_vector_type(16))) _Float16 v16h;
typedef __attribute__((ext_vector_type(8)))  float    v8f;

// ---------------- WMMA helpers ----------------
__device__ __forceinline__ int frag_k(int t, int grp) {
  // pair t in 0..7 -> starting K index of the two packed halves (16-bit A/B layout)
  return (t < 4 ? 2 * t : 16 + 2 * (t - 4)) + grp * 8;
}

// K-contiguous fragment load (two contiguous 16B chunks per lane -> b128 pairs)
__device__ __forceinline__ v16h load_frag_rowK(const _Float16* __restrict__ row, int grp) {
  v16h v;
#pragma unroll
  for (int t = 0; t < 8; ++t) {
    int k = frag_k(t, grp);
    v[2 * t]     = row[k];
    v[2 * t + 1] = row[k + 1];
  }
  return v;
}

__device__ __forceinline__ v8f wmma16(v16h a, v16h b, v8f c) {
  return __builtin_amdgcn_wmma_f32_16x16x32_f16(false, a, false, b, (short)0, c, false, false);
}

// ---------------- prep: pack weights to f16 GEMM layouts ----------------
__global__ void prep_kernel(const float* __restrict__ sigma, const float* __restrict__ Kbond,
                            const float* __restrict__ Knon, _Float16* __restrict__ Wh,
                            _Float16* __restrict__ KhH, _Float16* __restrict__ KhT,
                            _Float16* __restrict__ A1h, _Float16* __restrict__ W2h) {
  int tid = blockIdx.x * blockDim.x + threadIdx.x;
  int stride = gridDim.x * blockDim.x;
  // conv2d weights: Wh[co][k], k = ci*25+dy*5+dx, abs(), zero pad 400..415
  for (int i = tid; i < NCH * KC2D; i += stride) {
    int co = i / KC2D, k = i - co * KC2D;
    float v = 0.f;
    if (k < 400) {
      int ci = k / 25, rem = k - ci * 25;
      v = fabsf(sigma[((co * CIN + ci) * 5 + rem / 5) * 5 + rem % 5]);
    }
    Wh[i] = (_Float16)v;
  }
  // bonded Khat: KhH[l][f][d pad 128]   (B-matrix for gemm1: rows f, K=d contiguous)
  for (int i = tid; i < 3 * NFILT * 128; i += stride) {
    int l = i / (NFILT * 128), rem = i - l * NFILT * 128;
    int f = rem / 128, d = rem - f * 128;
    KhH[i] = (_Float16)((d < DFEAT) ? Kbond[(l * NFILT + f) * DFEAT + d] : 0.f);
  }
  // transposed Khat: KhT[l][d][f]       (B-matrix for gemm2: rows d, K=f contiguous)
  for (int i = tid; i < 3 * DFEAT * NFILT; i += stride) {
    int l = i / (DFEAT * NFILT), rem = i - l * DFEAT * NFILT;
    int d = rem / NFILT, f = rem - d * NFILT;
    KhT[i] = (_Float16)Kbond[(l * NFILT + f) * DFEAT + d];
  }
  // conv1 A: A1h[l][f][k=c*11+r pad 1248]
  for (int i = tid; i < 3 * NFILT * KCV1; i += stride) {
    int l = i / (NFILT * KCV1), rem = i - l * NFILT * KCV1;
    int f = rem / KCV1, k = rem - f * KCV1;
    float v = 0.f;
    if (k < 1232) {
      int c = k / 11, r = k - c * 11;
      v = Knon[((l * NFILT + f) * DFEAT + c) * 11 + r];
    }
    A1h[i] = (_Float16)v;
  }
  // conv_transpose A: W2h[l][c][k=f*11+rho]
  for (int i = tid; i < 3 * DFEAT * KCV2; i += stride) {
    int l = i / (DFEAT * KCV2), rem = i - l * DFEAT * KCV2;
    int c = rem / KCV2, k = rem - c * KCV2;
    int f = k / 11, rho = k - f * 11;
    W2h[i] = (_Float16)Knon[((l * NFILT + f) * DFEAT + c) * 11 + rho];
  }
}

// ---------------- pairwise atom distances -> Dm f16 [B,16,N,N] ----------------
__global__ void pairdist_kernel(const float* __restrict__ X, _Float16* __restrict__ Dmh) {
  size_t t = (size_t)blockIdx.x * blockDim.x + threadIdx.x;
  if (t >= (size_t)BATCH * NRES * NRES) return;
  int j = t & 511, i = (t >> 9) & 511, b = (int)(t >> 18);
  float xi[4][3], xj[4][3];
#pragma unroll
  for (int a = 0; a < 4; ++a)
#pragma unroll
    for (int c = 0; c < 3; ++c) {
      xi[a][c] = X[(((size_t)b * NRES + i) * 4 + a) * 3 + c];
      xj[a][c] = X[(((size_t)b * NRES + j) * 4 + a) * 3 + c];
    }
#pragma unroll
  for (int ai = 0; ai < 4; ++ai)
#pragma unroll
    for (int aj = 0; aj < 4; ++aj) {
      float d0 = xi[ai][0] - xj[aj][0], d1 = xi[ai][1] - xj[aj][1], d2 = xi[ai][2] - xj[aj][2];
      float d = sqrtf(fmaxf(d0 * d0 + d1 * d1 + d2 * d2, 0.f));
      Dmh[(((size_t)b * CIN + (ai * 4 + aj)) * NRES + i) * NRES + j] = (_Float16)d;
    }
}

__global__ void zero_kernel(float* __restrict__ p, int n) {
  int t = blockIdx.x * blockDim.x + threadIdx.x;
  if (t < n) p[t] = 0.f;
}

// ---------------- conv2d implicit GEMM (WMMA), two passes ----------------
// PASS 1: accumulate per-(b,ch) sum of squares of Z.
// PASS 2: Dd[b,ch,col] += relu(exp(-10*Z*rnorm) - bias) summed over rows.
template <int PASS>
__global__ __launch_bounds__(32) void conv2d_pass(const _Float16* __restrict__ Dmh,
                                                  const _Float16* __restrict__ Wh,
                                                  const float* __restrict__ rnorm,
                                                  const float* __restrict__ bias,
                                                  float* __restrict__ sumsq,
                                                  float* __restrict__ DdT) {
  __shared__ __align__(16) _Float16 sW[NCH * KC2D];
  __shared__ __align__(16) _Float16 sP[16 * KC2D];
  const int lane = threadIdx.x, grp = lane >> 4, hl = lane & 15;
  const int x0 = blockIdx.x * 16, y0 = blockIdx.y * 32, b = blockIdx.z;

  for (int i = lane; i < (NCH * KC2D) / 8; i += 32)
    reinterpret_cast<uint4*>(sW)[i] = reinterpret_cast<const uint4*>(Wh)[i];

  float accum[3][8];
#pragma unroll
  for (int mt = 0; mt < 3; ++mt)
#pragma unroll
    for (int r = 0; r < 8; ++r) accum[mt][r] = 0.f;

  for (int y = y0; y < y0 + 32; ++y) {
    for (int idx = lane; idx < 16 * KC2D; idx += 32) {
      int n = idx / KC2D, k = idx - n * KC2D;
      _Float16 v = (_Float16)0.f;
      if (k < 400) {
        int ci = k / 25, rem = k - ci * 25;
        int yy = y + rem / 5 - 2, xx = x0 + n + rem % 5 - 2;
        if (yy >= 0 && yy < NRES && xx >= 0 && xx < NRES)
          v = Dmh[(((size_t)b * CIN + ci) * NRES + yy) * NRES + xx];
      }
      sP[idx] = v;
    }
    __syncthreads();

    v8f acc[3] = {};
    const _Float16* prow = sP + hl * KC2D;
    for (int kk = 0; kk < 13; ++kk) {
      v16h bf = load_frag_rowK(prow + kk * 32, grp);
#pragma unroll
      for (int mt = 0; mt < 3; ++mt) {
        v16h af = load_frag_rowK(sW + (mt * 16 + hl) * KC2D + kk * 32, grp);
        acc[mt] = wmma16(af, bf, acc[mt]);
      }
    }
    __syncthreads();

#pragma unroll
    for (int mt = 0; mt < 3; ++mt)
#pragma unroll
      for (int r = 0; r < 8; ++r) {
        float v = acc[mt][r];
        if (PASS == 1) {
          accum[mt][r] += v * v;
        } else {
          int ch = mt * 16 + grp * 8 + r;
          float e = __expf(-10.f * v * rnorm[b * NCH + ch]) - bias[ch];
          accum[mt][r] += (e > 0.f) ? e : 0.f;
        }
      }
  }

#pragma unroll
  for (int mt = 0; mt < 3; ++mt)
#pragma unroll
    for (int r = 0; r < 8; ++r) {
      int ch = mt * 16 + grp * 8 + r;
      if (PASS == 1) {
        float v = accum[mt][r];
#pragma unroll
        for (int m = 8; m >= 1; m >>= 1) v += __shfl_xor(v, m, 32);
        if (hl == 0) atomicAdd(&sumsq[b * NCH + ch], v);
      } else {
        atomicAdd(&DdT[((size_t)b * NCH + ch) * NRES + x0 + hl], accum[mt][r]);
      }
    }
}

__global__ void rnorm_kernel(const float* __restrict__ sumsq, float* __restrict__ rnorm) {
  int t = blockIdx.x * blockDim.x + threadIdx.x;
  if (t < BATCH * NCH) rnorm[t] = 1.f / fmaxf(sqrtf(sumsq[t]), 1e-12f);
}

// normalize Dd over channels per (b,col); transpose to [B,N,48]
__global__ void ddnorm_kernel(const float* __restrict__ DdT, float* __restrict__ Dd) {
  int t = blockIdx.x * blockDim.x + threadIdx.x;
  if (t >= BATCH * NRES) return;
  int b = t >> 9, n = t & 511;
  float vals[NCH], nrm = 0.f;
#pragma unroll
  for (int f = 0; f < NCH; ++f) {
    vals[f] = DdT[((size_t)b * NCH + f) * NRES + n];
    nrm += vals[f] * vals[f];
  }
  float rn = 1.f / fmaxf(sqrtf(nrm), 1e-12f);
#pragma unroll
  for (int f = 0; f < NCH; ++f) Dd[((size_t)b * NRES + n) * NCH + f] = vals[f] * rn;
}

// stats for FD: s[n], u[f], w[f], S1, S2 per batch (fdst stride 98)
__global__ void fdstats_kernel(const float* __restrict__ Dd, float* __restrict__ sbuf,
                               float* __restrict__ fdst) {
  __shared__ float sS[NRES];
  int b = blockIdx.x, n = threadIdx.x;
  float s = 0.f;
  for (int f = 0; f < NCH; ++f) s += Dd[((size_t)b * NRES + n) * NCH + f];
  sbuf[b * NRES + n] = s;
  sS[n] = s;
  __syncthreads();
  if (n < NCH) {
    float u = 0.f, w = 0.f;
    for (int m = 0; m < NRES; ++m) {
      float v = Dd[((size_t)b * NRES + m) * NCH + n];
      u += v;
      w += sS[m] * v;
    }
    fdst[b * 98 + n] = u;
    fdst[b * 98 + 48 + n] = w;
  }
  if (n == 0) {
    float S1 = 0.f, S2 = 0.f;
    for (int m = 0; m < NRES; ++m) { S1 += sS[m]; S2 += sS[m] * sS[m]; }
    fdst[b * 98 + 96] = S1;
    fdst[b * 98 + 97] = S2;
  }
}

// Fh = concat(normalize_n(FD), FS) with closed-form column norm
__global__ void fdfill_kernel(const float* __restrict__ sbuf, const float* __restrict__ fdst,
                              const float* __restrict__ FS, float* __restrict__ Fh) {
  int t = blockIdx.x * blockDim.x + threadIdx.x;
  if (t >= BATCH * NRES * DFEAT) return;
  int d = t % DFEAT, n = (t / DFEAT) & 511, b = t / (DFEAT * NRES);
  if (d < NCH) {
    float u = fdst[b * 98 + d], w = fdst[b * 98 + 48 + d];
    float S1 = fdst[b * 98 + 96], S2 = fdst[b * 98 + 97];
    float num = C1C * sbuf[b * NRES + n] * u + C2C * w;
    float n2 = C1C * C1C * u * u * S2 + 2.f * C1C * C2C * u * w * S1 +
               (float)NRES * C2C * C2C * w * w;
    Fh[t] = num / fmaxf(sqrtf(n2), 1e-12f);
  } else {
    Fh[t] = FS[((size_t)b * NRES + n) * EMBW + (d - NCH)];
  }
}

// per-layer reductions: s = rowsum(Fh), u = colsum(Fh), w = s^T Fh
__global__ void layer_reduce1_kernel(const float* __restrict__ Fh, float* __restrict__ sL,
                                     float* __restrict__ uL, float* __restrict__ wL) {
  __shared__ float sS[NRES];
  int b = blockIdx.x, n = threadIdx.x;
  float s = 0.f;
  for (int d = 0; d < DFEAT; ++d) s += Fh[((size_t)b * NRES + n) * DFEAT + d];
  sL[b * NRES + n] = s;
  sS[n] = s;
  __syncthreads();
  if (n < DFEAT) {
    float u = 0.f, w = 0.f;
    for (int m = 0; m < NRES; ++m) {
      float v = Fh[((size_t)b * NRES + m) * DFEAT + n];
      u += v;
      w += sS[m] * v;
    }
    uL[b * DFEAT + n] = u;
    wL[b * DFEAT + n] = w;
  }
}

// P (padded to 128/row) = f16(s*Fh); Q (raw [B,N,d] flat) = c1*s_n*u_d + c2*w_d
__global__ void qp_kernel(const float* __restrict__ Fh, const float* __restrict__ sL,
                          const float* __restrict__ uL, const float* __restrict__ wL,
                          _Float16* __restrict__ Ph, float* __restrict__ Qflat) {
  int t = blockIdx.x * blockDim.x + threadIdx.x;
  if (t >= BATCH * NRES * 128) return;
  int d = t & 127, n = (t >> 7) & 511, b = t >> 16;
  if (d < DFEAT) {
    int fidx = ((b * NRES + n) * DFEAT) + d;
    float sv = sL[b * NRES + n];
    Ph[t] = (_Float16)(sv * Fh[fidx]);
    Qflat[fidx] = C1C * sv * uL[b * DFEAT + d] + C2C * wL[b * DFEAT + d];
  } else {
    Ph[t] = (_Float16)0.f;  // zero pad keeps GEMM A loads unguarded
  }
}

// y = lrelu(P @ Khat^T)  -> Yh f16 [B,512,128]
__global__ __launch_bounds__(32) void gemm1_kernel(const _Float16* __restrict__ Ph,
                                                   const _Float16* __restrict__ KhL,
                                                   _Float16* __restrict__ Yh) {
  const int lane = threadIdx.x, grp = lane >> 4, hl = lane & 15;
  const int mt = blockIdx.x, ft = blockIdx.y, b = blockIdx.z;
  const _Float16* arow = Ph + ((size_t)b * NRES + mt * 16 + hl) * 128;
  const _Float16* brow = KhL + (ft * 16 + hl) * 128;
  v8f acc = {};
#pragma unroll
  for (int kk = 0; kk < 4; ++kk) {
    v16h af = load_frag_rowK(arow + kk * 32, grp);
    v16h bf = load_frag_rowK(brow + kk * 32, grp);
    acc = wmma16(af, bf, acc);
  }
#pragma unroll
  for (int r = 0; r < 8; ++r) {
    int nr = mt * 16 + grp * 8 + r;
    float v = acc[r];
    v = (v > 0.f) ? v : 0.2f * v;
    Yh[((size_t)b * NRES + nr) * NFILT + ft * 16 + hl] = (_Float16)v;
  }
}

// Fhb = s * (Y @ Khat), last residue zeroed  -> f32 [B,512,112]
__global__ __launch_bounds__(32) void gemm2_kernel(const _Float16* __restrict__ Yh,
                                                   const _Float16* __restrict__ KhTL,
                                                   const float* __restrict__ sL,
                                                   float* __restrict__ Fhb) {
  const int lane = threadIdx.x, grp = lane >> 4, hl = lane & 15;
  const int mt = blockIdx.x, dt = blockIdx.y, b = blockIdx.z;
  const _Float16* arow = Yh + ((size_t)b * NRES + mt * 16 + hl) * NFILT;
  const _Float16* brow = KhTL + (dt * 16 + hl) * NFILT;  // KhT[d][f], K=f contiguous
  v8f acc = {};
#pragma unroll
  for (int kk = 0; kk < 4; ++kk) {
    v16h af = load_frag_rowK(arow + kk * 32, grp);
    v16h bf = load_frag_rowK(brow + kk * 32, grp);
    acc = wmma16(af, bf, acc);
  }
  const int d = dt * 16 + hl;
#pragma unroll
  for (int r = 0; r < 8; ++r) {
    int nr = mt * 16 + grp * 8 + r;
    float v = (nr == NRES - 1) ? 0.f : sL[b * NRES + nr] * acc[r];
    Fhb[((size_t)b * NRES + nr) * DFEAT + d] = v;
  }
}

// y2 = lrelu(conv1d(Qr, Knon))  M=filters, N=positions, K=112*11
__global__ __launch_bounds__(32) void conv1_kernel(const float* __restrict__ Qflat,
                                                   const _Float16* __restrict__ A1L,
                                                   _Float16* __restrict__ Y2h) {
  __shared__ __align__(16) _Float16 sP[16 * KCV1];
  const int lane = threadIdx.x, grp = lane >> 4, hl = lane & 15;
  const int t0 = blockIdx.x * 16, b = blockIdx.y;
  for (int idx = lane; idx < 16 * KCV1; idx += 32) {
    int n = idx / KCV1, k = idx - n * KCV1;
    _Float16 v = (_Float16)0.f;
    if (k < 1232) {
      int c = k / 11, r = k - c * 11;
      int tt = t0 + n + r;
      if (tt < NRES) v = (_Float16)Qflat[(size_t)b * NRES * DFEAT + c * NRES + tt];
    }
    sP[idx] = v;
  }
  __syncthreads();
  v8f acc[8] = {};
  const _Float16* prow = sP + hl * KCV1;
  for (int kk = 0; kk < 39; ++kk) {
    v16h bf = load_frag_rowK(prow + kk * 32, grp);
#pragma unroll
    for (int ft = 0; ft < 8; ++ft) {
      v16h af = load_frag_rowK(A1L + (ft * 16 + hl) * KCV1 + kk * 32, grp);
      acc[ft] = wmma16(af, bf, acc[ft]);
    }
  }
  const int t = t0 + hl;
  if (t < 502) {
#pragma unroll
    for (int ft = 0; ft < 8; ++ft)
#pragma unroll
      for (int r = 0; r < 8; ++r) {
        int f = ft * 16 + grp * 8 + r;
        float v = acc[ft][r];
        v = (v > 0.f) ? v : 0.2f * v;
        Y2h[((size_t)b * NFILT + f) * 502 + t] = (_Float16)v;
      }
  }
}

// Qt = conv_transpose1d(y2) == conv(flip W) : M=channels(112), N=t(512), K=128*11
__global__ __launch_bounds__(32) void conv2_kernel(const _Float16* __restrict__ Y2h,
                                                   const _Float16* __restrict__ W2L,
                                                   float* __restrict__ Qtflat) {
  __shared__ __align__(16) _Float16 sP[16 * KCV2];
  const int lane = threadIdx.x, grp = lane >> 4, hl = lane & 15;
  const int t0 = blockIdx.x * 16, b = blockIdx.y;
  for (int idx = lane; idx < 16 * KCV2; idx += 32) {
    int n = idx / KCV2, k = idx - n * KCV2;
    int f = k / 11, rho = k - f * 11;
    int tt = t0 + n - rho;
    sP[idx] = (tt >= 0 && tt < 502) ? Y2h[((size_t)b * NFILT + f) * 502 + tt] : (_Float16)0.f;
  }
  __syncthreads();
  v8f acc[7] = {};
  const _Float16* prow = sP + hl * KCV2;
  for (int kk = 0; kk < 44; ++kk) {
    v16h bf = load_frag_rowK(prow + kk * 32, grp);
#pragma unroll
    for (int ct = 0; ct < 7; ++ct) {
      v16h af = load_frag_rowK(W2L + (ct * 16 + hl) * KCV2 + kk * 32, grp);
      acc[ct] = wmma16(af, bf, acc[ct]);
    }
  }
  const int t = t0 + hl;
#pragma unroll
  for (int ct = 0; ct < 7; ++ct)
#pragma unroll
    for (int r = 0; r < 8; ++r) {
      int c = ct * 16 + grp * 8 + r;
      Qtflat[(size_t)b * NRES * DFEAT + c * NRES + t] = acc[ct][r];
    }
}

// uq/wq over Qt (read as raw-reshaped [B,N,112]); also zero colnorm
__global__ void layer_reduce2_kernel(const float* __restrict__ Qtflat,
                                     const float* __restrict__ sL, float* __restrict__ uqL,
                                     float* __restrict__ wqL, float* __restrict__ cn) {
  int b = blockIdx.x, n = threadIdx.x;
  if (n < DFEAT) {
    float uq = 0.f, wq = 0.f;
    for (int m = 0; m < NRES; ++m) {
      float v = Qtflat[(size_t)b * NRES * DFEAT + m * DFEAT + n];
      uq += v;
      wq += sL[b * NRES + m] * v;
    }
    uqL[b * DFEAT + n] = uq;
    wqL[b * DFEAT + n] = wq;
    cn[b * DFEAT + n] = 0.f;
  }
}

__global__ void update_kernel(const float* __restrict__ Fh, const float* __restrict__ Fhb,
                              const float* __restrict__ sL, const float* __restrict__ uqL,
                              const float* __restrict__ wqL, float* __restrict__ Tmp,
                              float* __restrict__ cn) {
  int t = blockIdx.x * blockDim.x + threadIdx.x;
  if (t >= BATCH * NRES * DFEAT) return;
  int d = t % DFEAT, n = (t / DFEAT) & 511, b = t / (DFEAT * NRES);
  float fhub = C1C * sL[b * NRES + n] * uqL[b * DFEAT + d] + C2C * wqL[b * DFEAT + d];
  float v = Fh[t] - ALPHA_C * fhub - ALPHA_C * Fhb[t];
  Tmp[t] = v;
  atomicAdd(&cn[b * DFEAT + d], v * v);
}

__global__ void scale_kernel(const float* __restrict__ Tmp, const float* __restrict__ cn,
                             float* __restrict__ Fh) {
  int t = blockIdx.x * blockDim.x + threadIdx.x;
  if (t >= BATCH * NRES * DFEAT) return;
  int d = t % DFEAT, b = t / (DFEAT * NRES);
  Fh[t] = Tmp[t] / fmaxf(sqrtf(cn[b * DFEAT + d]), 1e-12f);
}

__global__ void energy_kernel(const float* __restrict__ Fh, float* __restrict__ out, int which) {
  __shared__ float red[256];
  int b = blockIdx.x, tid = threadIdx.x;
  float s = 0.f;
  for (int i = tid; i < NRES * DFEAT; i += 256) {
    float v = Fh[(size_t)b * NRES * DFEAT + i];
    s += v * v;
  }
  red[tid] = s;
  __syncthreads();
  for (int m = 128; m >= 1; m >>= 1) {
    if (tid < m) red[tid] += red[tid + m];
    __syncthreads();
  }
  if (tid == 0) out[b * 2 + which] = red[0];
}

// ---------------- host side ----------------
extern "C" void kernel_launch(void* const* d_in, const int* in_sizes, int n_in, void* d_out,
                              int out_size, void* d_ws, size_t ws_size, hipStream_t stream) {
  (void)in_sizes; (void)n_in; (void)out_size; (void)ws_size;
  const float* Xdecoy = (const float*)d_in[0];
  const float* Xnat   = (const float*)d_in[1];
  const float* embN   = (const float*)d_in[2];
  const float* embD   = (const float*)d_in[3];
  const float* sigma  = (const float*)d_in[4];
  const float* bias   = (const float*)d_in[5];
  const float* Kbond  = (const float*)d_in[6];
  const float* Knon   = (const float*)d_in[7];
  float* out = (float*)d_out;

  char* base = (char*)d_ws;
  size_t off = 0;
  auto carve = [&](size_t bytes) -> void* {
    off = (off + 255) & ~(size_t)255;
    void* p = base + off;
    off += bytes;
    return p;
  };
  _Float16* Wh   = (_Float16*)carve((size_t)NCH * KC2D * 2);
  _Float16* KhH  = (_Float16*)carve((size_t)3 * NFILT * 128 * 2);
  _Float16* KhT  = (_Float16*)carve((size_t)3 * DFEAT * NFILT * 2);
  _Float16* A1h  = (_Float16*)carve((size_t)3 * NFILT * KCV1 * 2);
  _Float16* W2h  = (_Float16*)carve((size_t)3 * DFEAT * KCV2 * 2);
  _Float16* Dmh  = (_Float16*)carve((size_t)BATCH * CIN * NRES * NRES * 2);
  float* sumsq   = (float*)carve((size_t)BATCH * NCH * 4);
  float* rnorm   = (float*)carve((size_t)BATCH * NCH * 4);
  float* DdT     = (float*)carve((size_t)BATCH * NCH * NRES * 4);
  float* Dd      = (float*)carve((size_t)BATCH * NRES * NCH * 4);
  float* sbuf    = (float*)carve((size_t)BATCH * NRES * 4);
  float* fdst    = (float*)carve((size_t)BATCH * 98 * 4);
  float* Fh      = (float*)carve((size_t)BATCH * NRES * DFEAT * 4);
  _Float16* Ph   = (_Float16*)carve((size_t)BATCH * NRES * 128 * 2);  // padded rows
  float* sL      = (float*)carve((size_t)BATCH * NRES * 4);
  float* uL      = (float*)carve((size_t)BATCH * DFEAT * 4);
  float* wL      = (float*)carve((size_t)BATCH * DFEAT * 4);
  float* Qflat   = (float*)carve((size_t)BATCH * NRES * DFEAT * 4);
  _Float16* Yh   = (_Float16*)carve((size_t)BATCH * NRES * NFILT * 2);
  float* Fhb     = (float*)carve((size_t)BATCH * NRES * DFEAT * 4);
  _Float16* Y2h  = (_Float16*)carve((size_t)BATCH * NFILT * 502 * 2);
  float* Qtflat  = (float*)carve((size_t)BATCH * NRES * DFEAT * 4);
  float* uqL     = (float*)carve((size_t)BATCH * DFEAT * 4);
  float* wqL     = (float*)carve((size_t)BATCH * DFEAT * 4);
  float* cn      = (float*)carve((size_t)BATCH * DFEAT * 4);
  float* Tmp     = (float*)carve((size_t)BATCH * NRES * DFEAT * 4);

  prep_kernel<<<128, 256, 0, stream>>>(sigma, Kbond, Knon, Wh, KhH, KhT, A1h, W2h);

  const int NE = BATCH * NRES * DFEAT;   // 229376
  const int NEP = BATCH * NRES * 128;    // 262144 (padded P)
  for (int which = 0; which < 2; ++which) {
    const float* X  = which ? Xnat : Xdecoy;
    const float* FS = which ? embN : embD;

    pairdist_kernel<<<4096, 256, 0, stream>>>(X, Dmh);
    zero_kernel<<<(BATCH * NCH + 255) / 256, 256, 0, stream>>>(sumsq, BATCH * NCH);
    zero_kernel<<<(BATCH * NCH * NRES + 255) / 256, 256, 0, stream>>>(DdT, BATCH * NCH * NRES);

    conv2d_pass<1><<<dim3(32, 16, BATCH), 32, 0, stream>>>(Dmh, Wh, rnorm, bias, sumsq, DdT);
    rnorm_kernel<<<1, 192, 0, stream>>>(sumsq, rnorm);
    conv2d_pass<2><<<dim3(32, 16, BATCH), 32, 0, stream>>>(Dmh, Wh, rnorm, bias, sumsq, DdT);

    ddnorm_kernel<<<(BATCH * NRES + 255) / 256, 256, 0, stream>>>(DdT, Dd);
    fdstats_kernel<<<BATCH, NRES, 0, stream>>>(Dd, sbuf, fdst);
    fdfill_kernel<<<(NE + 255) / 256, 256, 0, stream>>>(sbuf, fdst, FS, Fh);

    for (int l = 0; l < 3; ++l) {
      const _Float16* KhL  = KhH + (size_t)l * NFILT * 128;
      const _Float16* KhTL = KhT + (size_t)l * DFEAT * NFILT;
      const _Float16* A1L  = A1h + (size_t)l * NFILT * KCV1;
      const _Float16* W2L  = W2h + (size_t)l * DFEAT * KCV2;
      layer_reduce1_kernel<<<BATCH, NRES, 0, stream>>>(Fh, sL, uL, wL);
      qp_kernel<<<(NEP + 255) / 256, 256, 0, stream>>>(Fh, sL, uL, wL, Ph, Qflat);
      gemm1_kernel<<<dim3(32, 8, BATCH), 32, 0, stream>>>(Ph, KhL, Yh);
      gemm2_kernel<<<dim3(32, 7, BATCH), 32, 0, stream>>>(Yh, KhTL, sL, Fhb);
      conv1_kernel<<<dim3(32, BATCH), 32, 0, stream>>>(Qflat, A1L, Y2h);
      conv2_kernel<<<dim3(32, BATCH), 32, 0, stream>>>(Y2h, W2L, Qtflat);
      layer_reduce2_kernel<<<BATCH, NRES, 0, stream>>>(Qtflat, sL, uqL, wqL, cn);
      update_kernel<<<(NE + 255) / 256, 256, 0, stream>>>(Fh, Fhb, sL, uqL, wqL, Tmp, cn);
      scale_kernel<<<(NE + 255) / 256, 256, 0, stream>>>(Tmp, cn, Fh);
    }
    energy_kernel<<<BATCH, 256, 0, stream>>>(Fh, out, which);
  }
}
